// DisentangledSelfAttention_57973468561811
// MI455X (gfx1250) — matrix operand order repaired and checked by reading, not compiled, not run on hardware
//
#include <hip/hip_runtime.h>
#include <hip/hip_bf16.h>
#include <stdint.h>

// ---------------------------------------------------------------------------
// DeBERTa-style disentangled self-attention for MI455X (gfx1250, wave32).
// f16 WMMA (v_wmma_f32_16x16x32_f16) for all matmuls, fp32 accumulation,
// flash-style streaming softmax, TDM (tensor_load_to_lds) for pos-table
// tiles, GLOBAL_LOAD_ASYNC_TO_LDS_B128 for K/V staging.
// ---------------------------------------------------------------------------

typedef __attribute__((ext_vector_type(16))) _Float16 v16h;
typedef __attribute__((ext_vector_type(8)))  float    v8f;
typedef __attribute__((ext_vector_type(4)))  uint32_t u32x4;
typedef __attribute__((ext_vector_type(8)))  uint32_t u32x8;
typedef __attribute__((ext_vector_type(4)))  int      i32x4;

#define AS1 __attribute__((address_space(1)))
#define AS3 __attribute__((address_space(3)))

#define S_LEN 2048
#define DMODEL 1024
#define NHEAD 16
#define DHEAD 64
#define NBH 32              // B * NHEAD
#define RTAB 1024           // 2*MAX_REL

#if __has_builtin(__builtin_amdgcn_global_load_async_to_lds_b128) && \
    __has_builtin(__builtin_amdgcn_s_wait_asynccnt)
#define USE_ASYNC_LDS 1
#else
#define USE_ASYNC_LDS 0
#endif

// ---- WMMA fragment index helpers (per CDNA5 ISA 7.12.2 tables) -------------
// A (16x32 f16): lane L holds row M = L&15; VGPR v holds K pair:
//   K = 2v + (v>=4 ? 8:0) + (L>=16 ? 8:0)  (+half)
__device__ __forceinline__ int kA(int v, int lane) {
    return 2 * v + ((v >= 4) ? 8 : 0) + ((lane >= 16) ? 8 : 0);
}
// B (32x16 f16): lane L holds col N = L&15; K = 2v + (v>=4 ? 8:0) + (L>=16 ? 8:0) (+half)
__device__ __forceinline__ int kB(int v, int lane) {
    return 2 * v + ((v >= 4) ? 8 : 0) + ((lane >= 16) ? 8 : 0);
}

__device__ __forceinline__ v8f wmma_f16(v16h a, v16h b, v8f c) {
    return __builtin_amdgcn_wmma_f32_16x16x32_f16(
        /*neg_a=*/false, a, /*neg_b=*/false, b,
        /*c_mod=*/(short)0, c, /*reuse_a=*/false, /*reuse_b=*/false);
}

union FragU { v16h v; _Float16 h[16]; uint32_t u[8]; };

// Load A fragment from row-major f16 [.., ld], rows r0.., cols k0..k0+31.
__device__ __forceinline__ v16h load_a_frag(const _Float16* p, int ld, int r0,
                                            int k0, int lane) {
    FragU f;
    int m = lane & 15;
    const _Float16* row = p + (long)(r0 + m) * ld + k0;
#pragma unroll
    for (int v = 0; v < 8; ++v)
        f.u[v] = *(const uint32_t*)((const char*)row + 2 * kA(v, lane));
    return f.v;
}

// Load B fragment for C = A * W^T: B[k][n] = W[n][k]; W row-major [N, ld].
__device__ __forceinline__ v16h load_b_frag(const _Float16* p, int ld, int n0,
                                            int k0, int lane) {
    FragU f;
    int n = lane & 15;
    const _Float16* row = p + (long)(n0 + n) * ld + k0;
#pragma unroll
    for (int v = 0; v < 8; ++v)
        f.u[v] = *(const uint32_t*)((const char*)row + 2 * kB(v, lane));
    return f.v;
}

// Load B fragment where memory is [K rows][N cols] (K-major): B[k][n] = p[k][n].
__device__ __forceinline__ v16h load_bT_frag(const _Float16* p, int ld, int n0,
                                             int lane) {
    FragU f;
    int n = n0 + (lane & 15);
#pragma unroll
    for (int v = 0; v < 8; ++v) {
        int k = kB(v, lane);
        f.h[2 * v]     = p[(long)k * ld + n];
        f.h[2 * v + 1] = p[(long)(k + 1) * ld + n];
    }
    return f.v;
}

// ---------------------------------------------------------------------------
// fp32 -> f16 conversion
__global__ void cvt_f32_to_f16_kernel(const float* __restrict__ src,
                                      _Float16* __restrict__ dst, int n) {
    int i = blockIdx.x * blockDim.x + threadIdx.x;
    if (i < n) dst[i] = (_Float16)src[i];
}

// ---------------------------------------------------------------------------
// WMMA GEMM: C[M,N] = A[M,K] @ W^T (+bias). One wave = 32x64 tile
// (2 A-fragments reuse 4 B-fragments -> 8 WMMAs per 32-deep k-step).
// mode 0: f16 output scattered to [B,H,S,Dh]   (QKV path)
// mode 1: fp32 output row-major [M,N]          (output projection)
__global__ __launch_bounds__(32)
void gemm32x64_kernel(const _Float16* __restrict__ A,
                      const _Float16* __restrict__ W,
                      const float* __restrict__ bias,
                      void* __restrict__ out,
                      int M, int N, int K, int mode) {
    int lane = threadIdx.x;
    int r0 = blockIdx.x * 32;
    int n0 = blockIdx.y * 64;
    v8f acc[2][4] = {};
    for (int k0 = 0; k0 < K; k0 += 32) {
        v16h a0 = load_a_frag(A, K, r0, k0, lane);
        v16h a1 = load_a_frag(A, K, r0 + 16, k0, lane);
#pragma unroll
        for (int nt = 0; nt < 4; ++nt) {
            v16h b = load_b_frag(W, K, n0 + nt * 16, k0, lane);
            acc[0][nt] = wmma_f16(a0, b, acc[0][nt]);
            acc[1][nt] = wmma_f16(a1, b, acc[1][nt]);
        }
    }
    int hl = (lane >= 16) ? 8 : 0;
    int cl = lane & 15;
#pragma unroll
    for (int mt = 0; mt < 2; ++mt) {
#pragma unroll
        for (int nt = 0; nt < 4; ++nt) {
#pragma unroll
            for (int v = 0; v < 8; ++v) {
                int m = r0 + mt * 16 + v + hl;
                int n = n0 + nt * 16 + cl;
                float val = acc[mt][nt][v] + (bias ? bias[n] : 0.0f);
                if (mode == 0) {
                    // scatter to [B, H, S, Dh] f16
                    int b = m >> 11, s = m & 2047, h = n >> 6, d = n & 63;
                    ((_Float16*)out)[(((long)(b * NHEAD + h)) * S_LEN + s) * DHEAD + d] =
                        (_Float16)val;
                } else {
                    ((float*)out)[(long)m * N + n] = val;
                }
            }
        }
    }
}

// ---------------------------------------------------------------------------
// qsum[bh][d] = sum_s Q[bh][s][d]   (reference's einsum sums q over l)
__global__ void qsum_kernel(const _Float16* __restrict__ Q,
                            float* __restrict__ qsum) {
    int idx = blockIdx.x * blockDim.x + threadIdx.x;  // 0 .. NBH*DHEAD-1
    int bh = idx >> 6, d = idx & 63;
    const _Float16* p = Q + ((long)bh * S_LEN) * DHEAD + d;
    float s = 0.f;
    for (int t = 0; t < S_LEN; ++t) s += (float)p[(long)t * DHEAD];
    qsum[idx] = s;
}

// c2p_tab[bh][r] = dot(qsum[bh], pos_table[r])
__global__ void c2p_kernel(const float* __restrict__ qsum,
                           const float* __restrict__ pos,
                           float* __restrict__ c2p) {
    int idx = blockIdx.x * blockDim.x + threadIdx.x;  // 0 .. NBH*RTAB-1
    int bh = idx >> 10, r = idx & 1023;
    const float* q = qsum + bh * DHEAD;
    const float* pt = pos + (long)r * DHEAD;
    float s = 0.f;
#pragma unroll 8
    for (int d = 0; d < DHEAD; ++d) s += q[d] * pt[d];
    c2p[idx] = s;
}

// ---------------------------------------------------------------------------
// Flash attention with disentangled bias. One wave per (bh, 16-row i-tile).
// bias[i,j] = c2p_tab[bh][clip(i-j+512)] + dot(K[j], pos_table[clip(2559-i-j)])
// K/V blocks staged via GLOBAL_LOAD_ASYNC_TO_LDS_B128 (ASYNCcnt);
// pos-table window staged via TDM tensor_load_to_lds (TENSORcnt), whose
// OOB-reads-return-zero semantics handle the window clamp for free.
__global__ __launch_bounds__(32)
void flash_attn_kernel(const _Float16* __restrict__ Qh,
                       const _Float16* __restrict__ Kh,
                       const _Float16* __restrict__ Vh,
                       const float* __restrict__ c2p_tab,
                       const _Float16* __restrict__ PTh,
                       _Float16* __restrict__ Out) {
    __shared__ __align__(16) _Float16 ksm[32 * DHEAD];
    __shared__ __align__(16) _Float16 vsm[32 * DHEAD];
    __shared__ __align__(16) _Float16 pts[48 * DHEAD];
    __shared__ __align__(16) float    gsm[32 * 48];
    __shared__ __align__(16) _Float16 psm[16 * 32];
    __shared__ __align__(16) float    c2ps[RTAB];

    int lane = threadIdx.x;
    int bid = blockIdx.x;
    int it = bid & 127;        // 128 i-tiles of 16 rows
    int bh = bid >> 7;         // 0..31
    int b = bh >> 4, h = bh & 15;
    int i0 = it * 16;
    int hl = (lane >= 16) ? 8 : 0;
    int cl = lane & 15;

    for (int t = lane; t < RTAB; t += 32) c2ps[t] = c2p_tab[(long)bh * RTAB + t];

    const _Float16* Qbase = Qh + ((long)bh * S_LEN) * DHEAD;
    const _Float16* Kbase = Kh + ((long)bh * S_LEN) * DHEAD;
    const _Float16* Vbase = Vh + ((long)bh * S_LEN) * DHEAD;

    v16h qa0 = load_a_frag(Qbase, DHEAD, i0, 0, lane);
    v16h qa1 = load_a_frag(Qbase, DHEAD, i0, 32, lane);

    float mrow[8], lrow[8];
#pragma unroll
    for (int v = 0; v < 8; ++v) { mrow[v] = -1e30f; lrow[v] = 0.f; }
    v8f oacc[4] = {};
    const float scale = 0.125f;  // 1/sqrt(64)

    for (int j0 = 0; j0 < S_LEN; j0 += 32) {
        __syncthreads();

        // ---- stage K,V blocks (32 rows x 64 halfs, contiguous 4KB each) ----
#if USE_ASYNC_LDS
        {
            const _Float16* gk = Kbase + (long)j0 * DHEAD + lane * 8;
            const _Float16* gv = Vbase + (long)j0 * DHEAD + lane * 8;
            _Float16* lk = ksm + lane * 8;
            _Float16* lv = vsm + lane * 8;
#pragma unroll
            for (int t = 0; t < 8; ++t) {
                __builtin_amdgcn_global_load_async_to_lds_b128(
                    (AS1 i32x4*)(gk + t * 256), (AS3 i32x4*)(lk + t * 256), 0, 0);
                __builtin_amdgcn_global_load_async_to_lds_b128(
                    (AS1 i32x4*)(gv + t * 256), (AS3 i32x4*)(lv + t * 256), 0, 0);
            }
        }
#else
        {
            const uint4* ks = (const uint4*)(Kbase + (long)j0 * DHEAD);
            const uint4* vs = (const uint4*)(Vbase + (long)j0 * DHEAD);
            uint4* kd = (uint4*)ksm;
            uint4* vd = (uint4*)vsm;
            for (int t = lane; t < 256; t += 32) { kd[t] = ks[t]; vd[t] = vs[t]; }
        }
#endif

        // ---- pos-table window for p2c via TDM: rows rmin..rmin+47 ----
        int base = 2559 - i0 - j0;
        int rmin = base - 46;
        rmin = rmin < 0 ? 0 : (rmin > 1023 ? 1023 : rmin);
        {
            uint64_t gaddr = (uint64_t)(uintptr_t)(PTh + (size_t)rmin * DHEAD);
            uint32_t laddr = (uint32_t)(size_t)(AS3 void*)pts;
            uint32_t tdim1 = (uint32_t)(RTAB - rmin);  // rows left; OOB rows read 0
            u32x4 g0;
            g0[0] = 1u;                                   // count=1, user desc
            g0[1] = laddr;                                // lds_addr
            g0[2] = (uint32_t)gaddr;                      // global_addr[31:0]
            g0[3] = (((uint32_t)(gaddr >> 32)) & 0x01FFFFFFu) | 0x80000000u; // [56:32]+type=2
            u32x8 g1;
            g1[0] = 0x00010000u;                          // data_size=1 (2 bytes)
            g1[1] = (uint32_t)DHEAD << 16;                // tensor_dim0 = 64
            g1[2] = (tdim1 & 0xFFFFu) << 16;              // tensor_dim1 lo
            g1[3] = ((uint32_t)DHEAD << 16) | (tdim1 >> 16); // tile_dim0=64 | tdim1 hi
            g1[4] = 48u;                                  // tile_dim1 = 48
            g1[5] = (uint32_t)DHEAD;                      // tensor_dim0_stride = 64
            g1[6] = 0u;
            g1[7] = 0u;
            asm volatile("tensor_load_to_lds %0, %1" :: "s"(g0), "s"(g1) : "memory");
        }

#if USE_ASYNC_LDS
        __builtin_amdgcn_s_wait_asynccnt(0);
#endif
        __builtin_amdgcn_s_wait_tensorcnt(0);
        __syncthreads();

        // --- scores: Q (16x64) @ K^T (64x32) -> 2 tiles ---
        v8f sacc[2];
#pragma unroll
        for (int jt = 0; jt < 2; ++jt) {
            v16h b0 = load_b_frag(ksm, DHEAD, jt * 16, 0, lane);
            v16h b1 = load_b_frag(ksm, DHEAD, jt * 16, 32, lane);
            v8f z = {};
            z = wmma_f16(qa1, b1, z);
            sacc[jt] = wmma_f16(qa0, b0, z);
        }

        // --- G[j, r] = K_blk (32x64) @ PT_window^T (64x48) ---
#pragma unroll
        for (int jt = 0; jt < 2; ++jt) {
            v16h ka0 = load_a_frag(ksm, DHEAD, jt * 16, 0, lane);
            v16h ka1 = load_a_frag(ksm, DHEAD, jt * 16, 32, lane);
#pragma unroll
            for (int rt = 0; rt < 3; ++rt) {
                v16h pb0 = load_b_frag(pts, DHEAD, rt * 16, 0, lane);
                v16h pb1 = load_b_frag(pts, DHEAD, rt * 16, 32, lane);
                v8f g = {};
                g = wmma_f16(ka1, pb1, g);
                g = wmma_f16(ka0, pb0, g);
#pragma unroll
                for (int v = 0; v < 8; ++v)
                    gsm[(jt * 16 + v + hl) * 48 + rt * 16 + cl] = g[v];
            }
        }
        __syncthreads();

        // --- bias + online softmax (rows spread over 16-lane halves) ---
        float p0[8], p1[8], sfv[8];
#pragma unroll
        for (int v = 0; v < 8; ++v) {
            int gi = i0 + v + hl;
            int gj0 = j0 + cl;
            int gj1 = j0 + 16 + cl;
            int rel0 = gi - gj0 + 512; rel0 = rel0 < 0 ? 0 : (rel0 > 1023 ? 1023 : rel0);
            int rel1 = gi - gj1 + 512; rel1 = rel1 < 0 ? 0 : (rel1 > 1023 ? 1023 : rel1);
            int rr0 = 2559 - gi - gj0; rr0 = rr0 < 0 ? 0 : (rr0 > 1023 ? 1023 : rr0);
            int rr1 = 2559 - gi - gj1; rr1 = rr1 < 0 ? 0 : (rr1 > 1023 ? 1023 : rr1);
            float s0 = sacc[0][v] * scale + c2ps[rel0] + gsm[cl * 48 + (rr0 - rmin)];
            float s1 = sacc[1][v] * scale + c2ps[rel1] + gsm[(16 + cl) * 48 + (rr1 - rmin)];
            float t = fmaxf(s0, s1);
#pragma unroll
            for (int off = 1; off < 16; off <<= 1)
                t = fmaxf(t, __shfl_xor(t, off, 16));
            float mnew = fmaxf(mrow[v], t);
            float sc = __expf(mrow[v] - mnew);
            float e0 = __expf(s0 - mnew);
            float e1 = __expf(s1 - mnew);
            float rs = e0 + e1;
#pragma unroll
            for (int off = 1; off < 16; off <<= 1)
                rs += __shfl_xor(rs, off, 16);
            lrow[v] = lrow[v] * sc + rs;
            mrow[v] = mnew;
            sfv[v] = sc;
            p0[v] = e0;
            p1[v] = e1;
        }
#pragma unroll
        for (int v = 0; v < 8; ++v) {
#pragma unroll
            for (int nt = 0; nt < 4; ++nt) oacc[nt][v] *= sfv[v];
            psm[(v + hl) * 32 + cl]      = (_Float16)p0[v];
            psm[(v + hl) * 32 + 16 + cl] = (_Float16)p1[v];
        }
        __syncthreads();

        // --- O += P (16x32) @ V_blk (32x64) ---
        v16h pa = load_a_frag(psm, 32, 0, 0, lane);
#pragma unroll
        for (int nt = 0; nt < 4; ++nt) {
            v16h vb = load_bT_frag(vsm, DHEAD, nt * 16, lane);
            oacc[nt] = wmma_f16(pa, vb, oacc[nt]);
        }
    }

    // normalize and write [B, S, H*Dh] f16 (row-major for the final GEMM)
#pragma unroll
    for (int nt = 0; nt < 4; ++nt) {
#pragma unroll
        for (int v = 0; v < 8; ++v) {
            int s = i0 + v + hl;
            int d = nt * 16 + cl;
            float val = oacc[nt][v] / lrow[v];
            Out[((long)(b * S_LEN + s)) * DMODEL + h * DHEAD + d] = (_Float16)val;
        }
    }
}

// ---------------------------------------------------------------------------
extern "C" void kernel_launch(void* const* d_in, const int* in_sizes, int n_in,
                              void* d_out, int out_size, void* d_ws, size_t ws_size,
                              hipStream_t stream) {
    (void)in_sizes; (void)n_in; (void)out_size; (void)ws_size;
    const float* hidden = (const float*)d_in[0];
    const float* Wq = (const float*)d_in[1];
    const float* bq = (const float*)d_in[2];
    const float* Wk = (const float*)d_in[3];
    const float* bk = (const float*)d_in[4];
    const float* Wv = (const float*)d_in[5];
    const float* bv = (const float*)d_in[6];
    const float* Wc = (const float*)d_in[7];
    const float* pos = (const float*)d_in[8];

    const int M = 2 * S_LEN;       // 4096
    const int D = DMODEL;          // 1024

    char* ws = (char*)d_ws;
    size_t off = 0;
    auto carve = [&](size_t bytes) -> char* {
        char* p = ws + off;
        off += (bytes + 255) & ~(size_t)255;
        return p;
    };
    _Float16* Xh   = (_Float16*)carve((size_t)M * D * 2);        // 8 MB
    _Float16* Wqh  = (_Float16*)carve((size_t)D * D * 2);        // 2 MB
    _Float16* Wkh  = (_Float16*)carve((size_t)D * D * 2);
    _Float16* Wvh  = (_Float16*)carve((size_t)D * D * 2);
    _Float16* Wch  = (_Float16*)carve((size_t)D * D * 2);
    _Float16* PTh  = (_Float16*)carve((size_t)RTAB * DHEAD * 2); // 128 KB
    _Float16* Qh   = (_Float16*)carve((size_t)M * D * 2);        // 8 MB
    _Float16* Kh   = (_Float16*)carve((size_t)M * D * 2);
    _Float16* Vh   = (_Float16*)carve((size_t)M * D * 2);
    float*    qsum = (float*)carve((size_t)NBH * DHEAD * 4);     // 8 KB
    float*    c2pt = (float*)carve((size_t)NBH * RTAB * 4);      // 128 KB
    _Float16* AttnOut = Xh;  // reuse: Xh dead after the QKV GEMMs

    // 1) fp32 -> f16 conversions
    auto cvt = [&](const float* s, _Float16* d, int n) {
        cvt_f32_to_f16_kernel<<<dim3((n + 255) / 256), dim3(256), 0, stream>>>(s, d, n);
    };
    cvt(hidden, Xh, M * D);
    cvt(Wq, Wqh, D * D);
    cvt(Wk, Wkh, D * D);
    cvt(Wv, Wvh, D * D);
    cvt(Wc, Wch, D * D);
    cvt(pos, PTh, RTAB * DHEAD);

    // 2) QKV projection GEMMs (x @ W^T + b), scatter into [B,H,S,Dh] f16
    dim3 ggrid(M / 32, D / 64);
    gemm32x64_kernel<<<ggrid, 32, 0, stream>>>(Xh, Wqh, bq, Qh, M, D, D, 0);
    gemm32x64_kernel<<<ggrid, 32, 0, stream>>>(Xh, Wkh, bk, Kh, M, D, D, 0);
    gemm32x64_kernel<<<ggrid, 32, 0, stream>>>(Xh, Wvh, bv, Vh, M, D, D, 0);

    // 3) qsum over sequence, then c2p lookup table (B*H x 1024)
    qsum_kernel<<<dim3((NBH * DHEAD) / 256), dim3(256), 0, stream>>>(Qh, qsum);
    c2p_kernel<<<dim3((NBH * RTAB) / 256), dim3(256), 0, stream>>>(qsum, pos, c2pt);

    // 4) flash attention with disentangled bias
    flash_attn_kernel<<<dim3(NBH * (S_LEN / 16)), 32, 0, stream>>>(
        Qh, Kh, Vh, c2pt, PTh, AttnOut);

    // 5) output projection (no bias), fp32 result
    gemm32x64_kernel<<<ggrid, 32, 0, stream>>>(AttnOut, Wch, nullptr,
                                               (float*)d_out, M, D, D, 1);
}
